// andps_images_67783173865565
// MI455X (gfx1250) — matrix operand order repaired and checked by reading, not compiled, not run on hardware
//
#include <hip/hip_runtime.h>
#include <hip/hip_bf16.h>
#include <stdint.h>

typedef __attribute__((ext_vector_type(16))) _Float16 v16h;
typedef __attribute__((ext_vector_type(8)))  float    v8f;

#define BSZ       8192
#define XDIM      4099
#define DSD       3
#define IMG       64
#define C1        5
#define C2        10
#define P1        30     // pooled conv1 spatial
#define P2        13     // pooled conv2 spatial
#define FEAT_RAW  1693   // 10*13*13 + 3
#define FEAT_PAD  1728   // 27 * 64, K padded for WMMA double-buffered loop
#define N1        512
#define N2        256
#define N3        128
#define N4        64
#define NEXP      16

// ---------------------------------------------------------------------------
// conv1 (1->5, 5x5, VALID) + relu + maxpool2 ; out: [B,5,30,30] f16
// ---------------------------------------------------------------------------
__global__ void conv1_relu_pool(const float* __restrict__ x,
                                const float* __restrict__ w,
                                const float* __restrict__ bias,
                                _Float16* __restrict__ act1) {
  int idx = blockIdx.x * blockDim.x + threadIdx.x;
  const int total = BSZ * C1 * P1 * P1;
  if (idx >= total) return;
  int px = idx % P1;
  int py = (idx / P1) % P1;
  int c  = (idx / (P1 * P1)) % C1;
  int b  = idx / (P1 * P1 * C1);
  const float* img = x + (size_t)b * XDIM + DSD;   // 64x64 image
  const float* wc  = w + c * 25;
  float mx = -3.4e38f;
  #pragma unroll
  for (int dy = 0; dy < 2; ++dy) {
    #pragma unroll
    for (int dx = 0; dx < 2; ++dx) {
      int oy = py * 2 + dy, ox = px * 2 + dx;      // conv coords 0..59
      float s = bias[c];
      #pragma unroll
      for (int ky = 0; ky < 5; ++ky) {
        const float* row = img + (size_t)(oy + ky) * IMG + ox;
        #pragma unroll
        for (int kx = 0; kx < 5; ++kx)
          s = fmaf(row[kx], wc[ky * 5 + kx], s);
      }
      mx = fmaxf(mx, s);
    }
  }
  act1[idx] = (_Float16)fmaxf(mx, 0.0f);           // relu(max) == max(relu)
}

// ---------------------------------------------------------------------------
// conv2 (5->10, 5x5) + relu + maxpool2, scattered into padded feature matrix
// feat: [B, FEAT_PAD] f16 ; feature index = c*169 + py*13 + px
// ---------------------------------------------------------------------------
__global__ void conv2_relu_pool(const _Float16* __restrict__ act1,
                                const float* __restrict__ w,
                                const float* __restrict__ bias,
                                _Float16* __restrict__ feat) {
  int idx = blockIdx.x * blockDim.x + threadIdx.x;
  const int total = BSZ * C2 * P2 * P2;
  if (idx >= total) return;
  int px = idx % P2;
  int py = (idx / P2) % P2;
  int c  = (idx / (P2 * P2)) % C2;
  int b  = idx / (P2 * P2 * C2);
  const _Float16* in = act1 + (size_t)b * (C1 * P1 * P1);
  const float* wc = w + c * (C1 * 25);
  float mx = -3.4e38f;
  #pragma unroll
  for (int dy = 0; dy < 2; ++dy) {
    #pragma unroll
    for (int dx = 0; dx < 2; ++dx) {
      int oy = py * 2 + dy, ox = px * 2 + dx;      // conv coords 0..25
      float s = bias[c];
      for (int ci = 0; ci < C1; ++ci) {
        const _Float16* ch = in + ci * (P1 * P1);
        const float* wci = wc + ci * 25;
        #pragma unroll
        for (int ky = 0; ky < 5; ++ky) {
          const _Float16* row = ch + (size_t)(oy + ky) * P1 + ox;
          #pragma unroll
          for (int kx = 0; kx < 5; ++kx)
            s = fmaf((float)row[kx], wci[ky * 5 + kx], s);
        }
      }
      mx = fmaxf(mx, s);
    }
  }
  feat[(size_t)b * FEAT_PAD + c * (P2 * P2) + py * P2 + px] =
      (_Float16)fmaxf(mx, 0.0f);
}

// append the 3 state dims at feature columns 1690..1692
__global__ void write_state(const float* __restrict__ x,
                            _Float16* __restrict__ feat) {
  int idx = blockIdx.x * blockDim.x + threadIdx.x;
  if (idx >= BSZ * DSD) return;
  int j = idx % DSD, b = idx / DSD;
  feat[(size_t)b * FEAT_PAD + (FEAT_RAW - DSD) + j] =
      (_Float16)x[(size_t)b * XDIM + j];
}

// f32 [N,Ks] -> f16 [N,Kd] with zero K-padding
__global__ void cvt_pad(const float* __restrict__ src,
                        _Float16* __restrict__ dst, int N, int Ks, int Kd) {
  int idx = blockIdx.x * blockDim.x + threadIdx.x;
  if (idx >= N * Kd) return;
  int k = idx % Kd, n = idx / Kd;
  float v = (k < Ks) ? src[(size_t)n * Ks + k] : 0.0f;
  dst[(size_t)n * Kd + k] = (_Float16)v;
}

// ---------------------------------------------------------------------------
// WMMA GEMM: D[M,N] = relu(A[M,K] * W[N,K]^T + bias[N]), f16 in/out, f32 acc
// One wave computes a 16x64 output tile (4 accumulators, A reused 4x).
// Double-buffered K loop, step 64 (two 16x16x32 WMMA groups per iter).
// Requirements: K % 64 == 0, N % 64 == 0, M % 128 == 0. 8 waves / block.
// ---------------------------------------------------------------------------
typedef union { v16h v; uint32_t u[8]; } frag_t;

__device__ __forceinline__ void load_a(frag_t& f, const uint32_t* p) {
  // regs 0..3: K pairs at p[0..3]; regs 4..7: K pairs at p[4..7]
  #pragma unroll
  for (int i = 0; i < 4; ++i) f.u[i] = p[i];
  #pragma unroll
  for (int i = 0; i < 4; ++i) f.u[4 + i] = p[4 + i];
}
__device__ __forceinline__ void load_b(frag_t& f, const uint32_t* p) {
  #pragma unroll
  for (int i = 0; i < 8; ++i) f.u[i] = p[i];
}

__global__ void gemm_bias_relu_wmma(const _Float16* __restrict__ A,
                                    const _Float16* __restrict__ W,
                                    const float* __restrict__ bias,
                                    _Float16* __restrict__ D,
                                    int M, int N, int K) {
  const int lane = threadIdx.x & 31;
  const int wave = threadIdx.x >> 5;
  const int m0 = (blockIdx.y * 8 + wave) * 16;
  const int n0 = blockIdx.x * 64;

  // A 16x32 layout (matches B): lanes 0-15 rows M=0..15, K 0..15;
  // lanes 16-31 same rows, K 16..31. Each reg = one f16 K-pair.
  const int aRow = m0 + (lane & 15);
  const int aKb  = (lane & 16) ? 16 : 0;
  // B 32x16: lane column n = lane&15; lanes 0-15 K=0..15, lanes 16-31 K=16..31.
  const int bRow = n0 + (lane & 15);               // W row == B column
  const int bKb  = (lane & 16) ? 16 : 0;

  const uint32_t* ap = (const uint32_t*)(A + (size_t)aRow * K) + (aKb >> 1);
  const uint32_t* bp0 = (const uint32_t*)(W + (size_t)(bRow +  0) * K) + (bKb >> 1);
  const uint32_t* bp1 = (const uint32_t*)(W + (size_t)(bRow + 16) * K) + (bKb >> 1);
  const uint32_t* bp2 = (const uint32_t*)(W + (size_t)(bRow + 32) * K) + (bKb >> 1);
  const uint32_t* bp3 = (const uint32_t*)(W + (size_t)(bRow + 48) * K) + (bKb >> 1);

  v8f acc0 = {}, acc1 = {}, acc2 = {}, acc3 = {};
  frag_t a0, a1, b00, b01, b02, b03, b10, b11, b12, b13;

  // prologue: fetch K-block 0 (half 0)
  load_a(a0, ap);
  load_b(b00, bp0); load_b(b01, bp1); load_b(b02, bp2); load_b(b03, bp3);

  for (int k = 0; k < K; k += 64) {
    const int h1 = (k + 32) >> 1;                  // uint32 offset of half 1
    // fetch half 1 while computing on half 0
    load_a(a1, ap + h1);
    load_b(b10, bp0 + h1); load_b(b11, bp1 + h1);
    load_b(b12, bp2 + h1); load_b(b13, bp3 + h1);

    acc0 = __builtin_amdgcn_wmma_f32_16x16x32_f16(false, a0.v, false, b00.v, (short)0, acc0, false, false);
    acc1 = __builtin_amdgcn_wmma_f32_16x16x32_f16(false, a0.v, false, b01.v, (short)0, acc1, false, false);
    acc2 = __builtin_amdgcn_wmma_f32_16x16x32_f16(false, a0.v, false, b02.v, (short)0, acc2, false, false);
    acc3 = __builtin_amdgcn_wmma_f32_16x16x32_f16(false, a0.v, false, b03.v, (short)0, acc3, false, false);

    if (k + 64 < K) {                              // fetch next half 0
      const int h0 = (k + 64) >> 1;
      load_a(a0, ap + h0);
      load_b(b00, bp0 + h0); load_b(b01, bp1 + h0);
      load_b(b02, bp2 + h0); load_b(b03, bp3 + h0);
    }

    acc0 = __builtin_amdgcn_wmma_f32_16x16x32_f16(false, a1.v, false, b10.v, (short)0, acc0, false, false);
    acc1 = __builtin_amdgcn_wmma_f32_16x16x32_f16(false, a1.v, false, b11.v, (short)0, acc1, false, false);
    acc2 = __builtin_amdgcn_wmma_f32_16x16x32_f16(false, a1.v, false, b12.v, (short)0, acc2, false, false);
    acc3 = __builtin_amdgcn_wmma_f32_16x16x32_f16(false, a1.v, false, b13.v, (short)0, acc3, false, false);
  }

  // C/D layout: reg r -> row m0 + r (+8 on upper lanes), col = n tile + lane&15
  const int cRow = m0 + ((lane & 16) ? 8 : 0);
  const int cCol = n0 + (lane & 15);
  v8f accs[4] = {acc0, acc1, acc2, acc3};
  #pragma unroll
  for (int j = 0; j < 4; ++j) {
    const int col = cCol + j * 16;
    const float bv = bias[col];
    #pragma unroll
    for (int r = 0; r < 8; ++r) {
      float v = fmaxf(accs[j][r] + bv, 0.0f);
      D[(size_t)(cRow + r) * N + col] = (_Float16)v;
    }
  }
}

// ---------------------------------------------------------------------------
// head: fc5 (64->16) + softmax + expert mix   out[b,i] = sum_n w_n (A_n diff)_i
// ---------------------------------------------------------------------------
__global__ void head_kernel(const _Float16* __restrict__ h4,
                            const float* __restrict__ w5,
                            const float* __restrict__ b5,
                            const float* __restrict__ Bm,
                            const float* __restrict__ Cm,
                            const float* __restrict__ xtar,
                            const float* __restrict__ x,
                            float* __restrict__ out) {
  int b = blockIdx.x * blockDim.x + threadIdx.x;
  if (b >= BSZ) return;
  float h[N4];
  const _Float16* hr = h4 + (size_t)b * N4;
  #pragma unroll
  for (int k = 0; k < N4; ++k) h[k] = (float)hr[k];

  float logit[NEXP];
  float mx = -3.4e38f;
  for (int n = 0; n < NEXP; ++n) {
    float s = b5[n];
    const float* wr = w5 + n * N4;
    #pragma unroll
    for (int k = 0; k < N4; ++k) s = fmaf(h[k], wr[k], s);
    logit[n] = s;
    mx = fmaxf(mx, s);
  }
  float sum = 0.0f;
  for (int n = 0; n < NEXP; ++n) {
    float e = expf(logit[n] - mx);
    logit[n] = e;
    sum += e;
  }
  const float inv = 1.0f / sum;

  float diff[DSD];
  #pragma unroll
  for (int j = 0; j < DSD; ++j)
    diff[j] = xtar[j] - x[(size_t)b * XDIM + j];

  float s0 = 0.f, s1 = 0.f, s2 = 0.f;
  for (int n = 0; n < NEXP; ++n) {
    float wn = logit[n] * inv;
    const float* Bn = Bm + n * 9;
    const float* Cn = Cm + n * 9;
    float r0 = (Bn[0]+Cn[0])*diff[0] + (Bn[1]+Cn[1])*diff[1] + (Bn[2]+Cn[2])*diff[2];
    float r1 = (Bn[3]+Cn[3])*diff[0] + (Bn[4]+Cn[4])*diff[1] + (Bn[5]+Cn[5])*diff[2];
    float r2 = (Bn[6]+Cn[6])*diff[0] + (Bn[7]+Cn[7])*diff[1] + (Bn[8]+Cn[8])*diff[2];
    s0 = fmaf(wn, r0, s0);
    s1 = fmaf(wn, r1, s1);
    s2 = fmaf(wn, r2, s2);
  }
  out[(size_t)b * DSD + 0] = s0;
  out[(size_t)b * DSD + 1] = s1;
  out[(size_t)b * DSD + 2] = s2;
}

// ---------------------------------------------------------------------------
extern "C" void kernel_launch(void* const* d_in, const int* in_sizes, int n_in,
                              void* d_out, int out_size, void* d_ws, size_t ws_size,
                              hipStream_t stream) {
  (void)in_sizes; (void)n_in; (void)out_size; (void)ws_size;
  const float* x    = (const float*)d_in[0];
  const float* c1w  = (const float*)d_in[1];
  const float* c1b  = (const float*)d_in[2];
  const float* c2w  = (const float*)d_in[3];
  const float* c2b  = (const float*)d_in[4];
  const float* f1w  = (const float*)d_in[5];
  const float* f1b  = (const float*)d_in[6];
  const float* f2w  = (const float*)d_in[7];
  const float* f2b  = (const float*)d_in[8];
  const float* f3w  = (const float*)d_in[9];
  const float* f3b  = (const float*)d_in[10];
  const float* f4w  = (const float*)d_in[11];
  const float* f4b  = (const float*)d_in[12];
  const float* f5w  = (const float*)d_in[13];
  const float* f5b  = (const float*)d_in[14];
  const float* Bm   = (const float*)d_in[15];
  const float* Cm   = (const float*)d_in[16];
  const float* xtar = (const float*)d_in[17];
  float* out = (float*)d_out;

  // workspace carve (256B aligned)
  char* p = (char*)d_ws;
  auto carve = [&](size_t bytes) {
    char* r = p;
    p += (bytes + 255) & ~(size_t)255;
    return r;
  };
  _Float16* act1 = (_Float16*)carve((size_t)BSZ * C1 * P1 * P1 * 2);  // 73.7 MB
  _Float16* feat = (_Float16*)carve((size_t)BSZ * FEAT_PAD * 2);      // 28.3 MB
  _Float16* h1   = (_Float16*)carve((size_t)BSZ * N1 * 2);
  _Float16* h2   = (_Float16*)carve((size_t)BSZ * N2 * 2);
  _Float16* h3   = (_Float16*)carve((size_t)BSZ * N3 * 2);
  _Float16* h4   = (_Float16*)carve((size_t)BSZ * N4 * 2);
  _Float16* w1h  = (_Float16*)carve((size_t)N1 * FEAT_PAD * 2);
  _Float16* w2h  = (_Float16*)carve((size_t)N2 * N1 * 2);
  _Float16* w3h  = (_Float16*)carve((size_t)N3 * N2 * 2);
  _Float16* w4h  = (_Float16*)carve((size_t)N4 * N3 * 2);

  // zero feature matrix (covers the K padding columns)
  hipMemsetAsync(feat, 0, (size_t)BSZ * FEAT_PAD * 2, stream);

  const int T = 256;
  auto blocks = [](long long n, int t) { return (unsigned)((n + t - 1) / t); };

  // weight conversions (f32 -> f16, K-padded where needed)
  cvt_pad<<<blocks((long long)N1 * FEAT_PAD, T), T, 0, stream>>>(f1w, w1h, N1, FEAT_RAW, FEAT_PAD);
  cvt_pad<<<blocks((long long)N2 * N1, T), T, 0, stream>>>(f2w, w2h, N2, N1, N1);
  cvt_pad<<<blocks((long long)N3 * N2, T), T, 0, stream>>>(f3w, w3h, N3, N2, N2);
  cvt_pad<<<blocks((long long)N4 * N3, T), T, 0, stream>>>(f4w, w4h, N4, N3, N3);

  // convolutional front end
  conv1_relu_pool<<<blocks((long long)BSZ * C1 * P1 * P1, T), T, 0, stream>>>(x, c1w, c1b, act1);
  conv2_relu_pool<<<blocks((long long)BSZ * C2 * P2 * P2, T), T, 0, stream>>>(act1, c2w, c2b, feat);
  write_state<<<blocks((long long)BSZ * DSD, T), T, 0, stream>>>(x, feat);

  // WMMA FC stack: grid (N/64, M/128), 8 waves per block, 16x64 tile per wave
  gemm_bias_relu_wmma<<<dim3(N1 / 64, BSZ / 128), T, 0, stream>>>(feat, w1h, f1b, h1, BSZ, N1, FEAT_PAD);
  gemm_bias_relu_wmma<<<dim3(N2 / 64, BSZ / 128), T, 0, stream>>>(h1, w2h, f2b, h2, BSZ, N2, N1);
  gemm_bias_relu_wmma<<<dim3(N3 / 64, BSZ / 128), T, 0, stream>>>(h2, w3h, f3b, h3, BSZ, N3, N2);
  gemm_bias_relu_wmma<<<dim3(N4 / 64, BSZ / 128), T, 0, stream>>>(h3, w4h, f4b, h4, BSZ, N4, N3);

  // head: fc5 + softmax + expert mix
  head_kernel<<<blocks(BSZ, T), T, 0, stream>>>(h4, f5w, f5b, Bm, Cm, xtar, x, out);
}
